// GlobalRouter_52201032515627
// MI455X (gfx1250) — matrix operand order repaired and checked by reading, not compiled, not run on hardware
//
#include <hip/hip_runtime.h>
#include <hip/hip_bf16.h>

typedef _Float16 v8h  __attribute__((ext_vector_type(8)));
typedef _Float16 v16h __attribute__((ext_vector_type(16)));
typedef float    v8f  __attribute__((ext_vector_type(8)));
typedef int      v4i  __attribute__((vector_size(16)));

#define WMMA_F16(A, B, C) \
    __builtin_amdgcn_wmma_f32_16x16x32_f16(false, (A), false, (B), (short)0, (C), false, false)

// ---------------------------------------------------------------------------
// CDNA5 async global->LDS copy (ASYNCcnt-tracked).  The builtin takes
// (v4i addrspace(1)*, v4i addrspace(3)*, imm offset, imm cpol); we reach the
// typed AS pointers via integer round-trips (global VA == AS1 value; flat LDS
// pointer low 32 bits == LDS byte offset == AS3 value).
// ---------------------------------------------------------------------------
#if defined(__has_builtin) && __has_builtin(__builtin_amdgcn_global_load_async_to_lds_b128)
#define ASYNC_CP(g, l)                                                              \
    __builtin_amdgcn_global_load_async_to_lds_b128(                                 \
        (__attribute__((address_space(1))) v4i*)(unsigned long long)(const void*)(g), \
        (__attribute__((address_space(3))) v4i*)(unsigned)(unsigned long long)(const void*)(l), \
        0, 0)
#else
#define ASYNC_CP(g, l)                                                              \
    asm volatile("global_load_async_to_lds_b128 %0, %1, off"                        \
                 :: "v"((unsigned)(unsigned long long)(const void*)(l)),            \
                    "v"((const void*)(g))                                           \
                 : "memory")
#endif

#if defined(__has_builtin) && __has_builtin(__builtin_amdgcn_s_wait_asynccnt)
#define WAIT_ASYNC0 __builtin_amdgcn_s_wait_asynccnt(0)
#else
#define WAIT_ASYNC0 asm volatile("s_wait_asynccnt 0" ::: "memory")
#endif

static __device__ __forceinline__ v16h cat8(v8h lo, v8h hi) {
    v16h r;
#pragma unroll
    for (int i = 0; i < 8; ++i) { r[i] = lo[i]; r[8 + i] = hi[i]; }
    return r;
}

// ---------------------------------------------------------------------------
// Problem constants
// ---------------------------------------------------------------------------
#define BB 4
#define SS 2048
#define DD 512
#define HH 8
#define HD 64
#define NN 64
#define MROWS (BB * SS)          // 8192
#define E3D   (3 * DD)           // 1536

// ---------------------------------------------------------------------------
// K0: f32 -> f16 convert
// ---------------------------------------------------------------------------
__global__ void k_cvt_f32_f16(const float* __restrict__ src, _Float16* __restrict__ dst, int n) {
    int i = blockIdx.x * blockDim.x + threadIdx.x;
    if (i < n) dst[i] = (_Float16)src[i];
}

// ---------------------------------------------------------------------------
// K1: QKV projection.  qkv[m, e] = sum_d xh[m,d] * w_in[e,d] + b[e]
// Per wave: 16 (M) x 64 (E) tile, K-loop over 512 in steps of 32.
// Scatter outputs: Q,K head-major [B,H,S,64]; V transposed [B,H,64,S] (all f16)
// ---------------------------------------------------------------------------
__global__ void k_gemm_qkv(const _Float16* __restrict__ xh,
                           const _Float16* __restrict__ w,
                           const float*    __restrict__ bias,
                           _Float16* __restrict__ Qh,
                           _Float16* __restrict__ Kh,
                           _Float16* __restrict__ Vt) {
    const int wid  = blockIdx.x * (blockDim.x >> 5) + (threadIdx.x >> 5);
    const int lane = threadIdx.x & 31;
    const int half = lane >> 4, l16 = lane & 15;
    const int NT = E3D / 64;                       // 24 n-tiles of 64
    const int mtile = wid / NT, ntile = wid % NT;
    const int m0 = mtile * 16, n0 = ntile * 64;
    const int arow = m0 + l16;

    v8f acc[4] = {};
#pragma unroll 4
    for (int kk = 0; kk < DD; kk += 32) {
        // A operand: lane<16 -> row l16, k {kk..kk+7, kk+16..23}; lane>=16 -> {+8..15, +24..31}
        const _Float16* ap = xh + (size_t)arow * DD + kk + half * 8;
        v16h A = cat8(*(const v8h*)ap, *(const v8h*)(ap + 16));
#pragma unroll
        for (int t = 0; t < 4; ++t) {
            const int col = n0 + t * 16 + l16;
            const _Float16* bp = w + (size_t)col * DD + kk + half * 8;
            v16h Bv = cat8(*(const v8h*)bp, *(const v8h*)(bp + 16));
            acc[t] = WMMA_F16(A, Bv, acc[t]);
        }
    }
#pragma unroll
    for (int t = 0; t < 4; ++t) {
        const int col = n0 + t * 16 + l16;
        const float bv = bias[col];
#pragma unroll
        for (int r = 0; r < 8; ++r) {
            const int row = m0 + r + 8 * half;      // global M row
            const int b = row >> 11, s = row & 2047;
            const _Float16 hv = (_Float16)(acc[t][r] + bv);
            if (col < DD) {
                const int h = col >> 6, hd = col & 63;
                Qh[(((size_t)(b * HH + h) * SS + s) << 6) + hd] = hv;
            } else if (col < 2 * DD) {
                const int e2 = col - DD, h = e2 >> 6, hd = e2 & 63;
                Kh[(((size_t)(b * HH + h) * SS + s) << 6) + hd] = hv;
            } else {
                const int e2 = col - 2 * DD, h = e2 >> 6, hd = e2 & 63;
                Vt[((size_t)(b * HH + h) * HD + hd) * SS + s] = hv;
            }
        }
    }
}

// ---------------------------------------------------------------------------
// K2: flash attention.  One wave = one (b,h, 16-row q-tile); the 4 waves of a
// block always share (b,h) (4 | 128 q-tiles), so K/V 32-key chunks are staged
// once per block into double-buffered LDS with async global->LDS copies
// (ASYNCcnt) overlapping the WMMA compute on the other buffer.
// ---------------------------------------------------------------------------
__global__ void k_attn(const _Float16* __restrict__ Qh,
                       const _Float16* __restrict__ Kh,
                       const _Float16* __restrict__ Vt,
                       _Float16* __restrict__ ctxh) {
    __shared__ __align__(16) _Float16 pbuf[4][16 * 32];      // per-wave P transpose
    __shared__ __align__(16) _Float16 kstage[2][32 * 64];    // [key][hd]
    __shared__ __align__(16) _Float16 vstage[2][64 * 32];    // [hd][key]
    const int tid  = threadIdx.x;
    const int wIn  = tid >> 5;
    const int wid  = blockIdx.x * 4 + wIn;
    const int lane = tid & 31;
    const int half = lane >> 4, l16 = lane & 15;
    const int qt = wid & 127;                  // S/16 = 128 q-tiles
    const int bh = wid >> 7;                   // 0..31 (uniform across block)
    const int q0 = qt * 16;
    const size_t baseQK = (size_t)bh * SS * HD;
    const size_t baseV  = (size_t)bh * HD * SS;

    // Q A-operands (held in registers for the whole key loop): k 0..31 and 32..63
    const _Float16* qp = Qh + baseQK + (size_t)(q0 + l16) * HD + half * 8;
    const v16h A0 = cat8(*(const v8h*)qp,        *(const v8h*)(qp + 16));
    const v16h A1 = cat8(*(const v8h*)(qp + 32), *(const v8h*)(qp + 48));

    v8f acc[4] = {};
    float rmax[8], rsum[8];
#pragma unroll
    for (int r = 0; r < 8; ++r) { rmax[r] = -1e30f; rsum[r] = 0.f; }
    _Float16* pw = &pbuf[wIn][0];

    // Cooperative staging split: 128 threads, 4 async b128 each per chunk.
    // K chunk: 32 rows x 64 halves, fully contiguous (4 KB).
    // V chunk: 64 hd-rows x 32 halves (row stride SS in global).
    const int vr = tid >> 1;            // hd row 0..63
    const int vc = (tid & 1) * 16;      // half-offset within 32-half row
    auto stage = [&](int kc, int sbuf) {
        const _Float16* Kg = Kh + baseQK + (size_t)kc * HD;
        _Float16* kl = &kstage[sbuf][0];
        ASYNC_CP(Kg + tid * 16,     kl + tid * 16);
        ASYNC_CP(Kg + tid * 16 + 8, kl + tid * 16 + 8);
        const _Float16* Vg = Vt + baseV + (size_t)vr * SS + kc + vc;
        _Float16* vl = &vstage[sbuf][vr * 32 + vc];
        ASYNC_CP(Vg,     vl);
        ASYNC_CP(Vg + 8, vl + 8);
    };

    stage(0, 0);
    WAIT_ASYNC0;
    __syncthreads();

    int buf = 0;
    for (int kc = 0; kc < SS; kc += 32) {
        if (kc + 32 < SS) stage(kc + 32, buf ^ 1);   // overlap with compute below
        const _Float16* kl = &kstage[buf][0];
        const _Float16* vl = &vstage[buf][0];

        // ---- scores: two 16x16 tiles (keys kc..+15 and kc+16..+31), K-dim = 64
        v8f sc[2];
#pragma unroll
        for (int t = 0; t < 2; ++t) {
            const _Float16* kp = kl + (t * 16 + l16) * 64 + half * 8;
            v16h B0 = cat8(*(const v8h*)kp,        *(const v8h*)(kp + 16));
            v16h B1 = cat8(*(const v8h*)(kp + 32), *(const v8h*)(kp + 48));
            v8f c = {};
            c = WMMA_F16(A0, B0, c);
            c = WMMA_F16(A1, B1, c);
            sc[t] = c;
        }
        // ---- online softmax over the 32-key chunk
#pragma unroll
        for (int r = 0; r < 8; ++r) {
            const float v0 = sc[0][r] * 0.125f;    // 1/sqrt(64)
            const float v1 = sc[1][r] * 0.125f;
            float cm = fmaxf(v0, v1);
            cm = fmaxf(cm, __shfl_xor(cm, 1));
            cm = fmaxf(cm, __shfl_xor(cm, 2));
            cm = fmaxf(cm, __shfl_xor(cm, 4));
            cm = fmaxf(cm, __shfl_xor(cm, 8));
            const float nm = fmaxf(rmax[r], cm);
            const float scale = __expf(rmax[r] - nm);
            rmax[r] = nm;
            const float p0 = __expf(v0 - nm);
            const float p1 = __expf(v1 - nm);
            float ps = p0 + p1;
            ps += __shfl_xor(ps, 1);
            ps += __shfl_xor(ps, 2);
            ps += __shfl_xor(ps, 4);
            ps += __shfl_xor(ps, 8);
            rsum[r] = rsum[r] * scale + ps;
#pragma unroll
            for (int t = 0; t < 4; ++t) acc[t][r] *= scale;
            const int prow = r + 8 * half;
            pw[prow * 32 + l16]      = (_Float16)p0;
            pw[prow * 32 + 16 + l16] = (_Float16)p1;
        }
        // ---- P -> A-operand (lane transpose via LDS; same-wave DS ops are in order)
        const _Float16* pr = pw + l16 * 32 + half * 8;
        const v16h AP = cat8(*(const v8h*)pr, *(const v8h*)(pr + 16));
        // ---- ctx += P @ V : four hd-tiles of 16
#pragma unroll
        for (int t = 0; t < 4; ++t) {
            const _Float16* vp = vl + (t * 16 + l16) * 32 + half * 8;
            v16h BV = cat8(*(const v8h*)vp, *(const v8h*)(vp + 16));
            acc[t] = WMMA_F16(AP, BV, acc[t]);
        }
        WAIT_ASYNC0;        // own async copies for next buffer complete
        __syncthreads();    // all waves' copies visible; all reads of buf done
        buf ^= 1;
    }
    // ---- normalize and scatter ctx (f16) to [B,S,D]
    const int b = bh >> 3, h = bh & 7;
#pragma unroll
    for (int t = 0; t < 4; ++t)
#pragma unroll
        for (int r = 0; r < 8; ++r) {
            const int q  = q0 + r + 8 * half;
            const int hd = t * 16 + l16;
            const float v = acc[t][r] / rsum[r];
            ctxh[((size_t)(b * SS + q) << 9) + h * HD + hd] = (_Float16)v;
        }
}

// ---------------------------------------------------------------------------
// K3: out projection. attended[m,e] = sum_d ctx[m,d]*w_out[e,d] + b[e]  (f32 out)
// ---------------------------------------------------------------------------
__global__ void k_gemm_out(const _Float16* __restrict__ ctxh,
                           const _Float16* __restrict__ w,
                           const float*    __restrict__ bias,
                           float* __restrict__ attended) {
    const int wid  = blockIdx.x * (blockDim.x >> 5) + (threadIdx.x >> 5);
    const int lane = threadIdx.x & 31;
    const int half = lane >> 4, l16 = lane & 15;
    const int NT = DD / 64;                        // 8
    const int mtile = wid / NT, ntile = wid % NT;
    const int m0 = mtile * 16, n0 = ntile * 64;
    const int arow = m0 + l16;

    v8f acc[4] = {};
#pragma unroll 4
    for (int kk = 0; kk < DD; kk += 32) {
        const _Float16* ap = ctxh + (size_t)arow * DD + kk + half * 8;
        v16h A = cat8(*(const v8h*)ap, *(const v8h*)(ap + 16));
#pragma unroll
        for (int t = 0; t < 4; ++t) {
            const int col = n0 + t * 16 + l16;
            const _Float16* bp = w + (size_t)col * DD + kk + half * 8;
            v16h Bv = cat8(*(const v8h*)bp, *(const v8h*)(bp + 16));
            acc[t] = WMMA_F16(A, Bv, acc[t]);
        }
    }
#pragma unroll
    for (int t = 0; t < 4; ++t) {
        const int col = n0 + t * 16 + l16;
        const float bv = bias[col];
#pragma unroll
        for (int r = 0; r < 8; ++r) {
            const int row = m0 + r + 8 * half;
            attended[(size_t)row * DD + col] = acc[t][r] + bv;
        }
    }
}

// ---------------------------------------------------------------------------
// K4: aff[m,n] = sum_d attended[m,d]*content_w[n,d] + b[n]   (N = 64)
// A operand converted f32->f16 on the fly.
// ---------------------------------------------------------------------------
__global__ void k_gemm_aff(const float* __restrict__ attended,
                           const _Float16* __restrict__ wc,
                           const float* __restrict__ bias,
                           float* __restrict__ aff) {
    const int wid  = blockIdx.x * (blockDim.x >> 5) + (threadIdx.x >> 5);
    const int lane = threadIdx.x & 31;
    const int half = lane >> 4, l16 = lane & 15;
    const int m0 = wid * 16;
    const int arow = m0 + l16;

    v8f acc[4] = {};
#pragma unroll 4
    for (int kk = 0; kk < DD; kk += 32) {
        const float* ap = attended + (size_t)arow * DD + kk + half * 8;
        v16h A;
#pragma unroll
        for (int i = 0; i < 8; ++i) { A[i] = (_Float16)ap[i]; A[8 + i] = (_Float16)ap[16 + i]; }
#pragma unroll
        for (int t = 0; t < 4; ++t) {
            const int col = t * 16 + l16;
            const _Float16* bp = wc + (size_t)col * DD + kk + half * 8;
            v16h Bv = cat8(*(const v8h*)bp, *(const v8h*)(bp + 16));
            acc[t] = WMMA_F16(A, Bv, acc[t]);
        }
    }
#pragma unroll
    for (int t = 0; t < 4; ++t) {
        const int col = t * 16 + l16;
        const float bv = bias[col];
#pragma unroll
        for (int r = 0; r < 8; ++r) {
            const int row = m0 + r + 8 * half;
            aff[(size_t)row * NN + col] = acc[t][r] + bv;
        }
    }
}

// ---------------------------------------------------------------------------
// K5: per-(b,n) sum & max over s of aff   (one block per (b,n))
// ---------------------------------------------------------------------------
__global__ void k_reduce(const float* __restrict__ aff,
                         float* __restrict__ sums, float* __restrict__ maxs) {
    const int b = blockIdx.x >> 6;
    const int n = blockIdx.x & 63;
    const int tid = threadIdx.x;
    float s = 0.f, m = -1e30f;
    for (int i = tid; i < SS; i += blockDim.x) {
        const float v = aff[((size_t)(b * SS + i)) * NN + n];
        s += v;
        m = fmaxf(m, v);
    }
    __shared__ float ss[256], sm[256];
    ss[tid] = s; sm[tid] = m;
    __syncthreads();
    for (int o = 128; o > 0; o >>= 1) {
        if (tid < o) { ss[tid] += ss[tid + o]; sm[tid] = fmaxf(sm[tid], sm[tid + o]); }
        __syncthreads();
    }
    if (tid == 0) { sums[b * NN + n] = ss[0]; maxs[b * NN + n] = sm[0]; }
}

// ---------------------------------------------------------------------------
// K6: final score, top-16, one-hot routing.  pos_imp is uniform 1/(S+1e-8)
// because softmax sums to 1 over keys, so weighted == sum_s(aff)/(S+1e-8).
// ---------------------------------------------------------------------------
__global__ void k_final(const float* __restrict__ sums, const float* __restrict__ maxs,
                        float* __restrict__ out_idx, float* __restrict__ out_rw) {
    const int b = blockIdx.x;
    const int n = threadIdx.x;          // 64 threads
    __shared__ float fin[NN];
    __shared__ int sel[16];
    const float s = sums[b * NN + n];
    const float m = maxs[b * NN + n];
    const float weighted = s / ((float)SS + 1e-8f);
    const float meanv = s * (1.0f / (float)SS);
    fin[n] = 0.5f * weighted + 0.3f * m + 0.2f * meanv;
    __syncthreads();
    if (n == 0) {
        float tmp[NN];
        for (int i = 0; i < NN; ++i) tmp[i] = fin[i];
        for (int j = 0; j < 16; ++j) {
            int bi = 0; float bvv = tmp[0];
            for (int i = 1; i < NN; ++i)
                if (tmp[i] > bvv) { bvv = tmp[i]; bi = i; }
            sel[j] = bi;
            tmp[bi] = -1e30f;
            out_idx[b * 16 + j] = (float)bi;
        }
    }
    __syncthreads();
    float rw = 0.f;
    for (int j = 0; j < 16; ++j)
        if (sel[j] == n) rw = 1.f;
    out_rw[b * NN + n] = rw;
}

// ---------------------------------------------------------------------------
// Launcher
// ---------------------------------------------------------------------------
extern "C" void kernel_launch(void* const* d_in, const int* in_sizes, int n_in,
                              void* d_out, int out_size, void* d_ws, size_t ws_size,
                              hipStream_t stream) {
    const float* x          = (const float*)d_in[0];   // [B,S,D]
    const float* in_proj_w  = (const float*)d_in[1];   // [3D,D]
    const float* in_proj_b  = (const float*)d_in[2];   // [3D]
    const float* out_proj_w = (const float*)d_in[3];   // [D,D]
    const float* out_proj_b = (const float*)d_in[4];   // [D]
    const float* content_w  = (const float*)d_in[5];   // [N,D]
    const float* content_b  = (const float*)d_in[6];   // [N]
    (void)in_sizes; (void)n_in; (void)out_size; (void)ws_size;

    // ---- workspace layout (f16 region, then f32 region) ----
    _Float16* wsh = (_Float16*)d_ws;
    const size_t nXH  = (size_t)MROWS * DD;            // 4,194,304
    const size_t nWIN = (size_t)E3D * DD;              //   786,432
    const size_t nWOUT= (size_t)DD * DD;               //   262,144
    const size_t nWC  = (size_t)NN * DD;               //    32,768
    const size_t nHead= (size_t)BB * HH * SS * HD;     // 4,194,304
    _Float16* xh    = wsh;                             // aliased by ctxh after K1
    _Float16* winh  = xh    + nXH;
    _Float16* wouth = winh  + nWIN;
    _Float16* wch   = wouth + nWOUT;
    _Float16* Qh    = wch   + nWC;
    _Float16* Kh    = Qh    + nHead;
    _Float16* Vt    = Kh    + nHead;
    _Float16* ctxh  = xh;                              // reuse xh (dead after K1)
    float* wsf  = (float*)(Vt + nHead);
    float* aff  = wsf;                                 // B*S*N = 524,288 floats
    float* sums = aff + (size_t)MROWS * NN;
    float* maxs = sums + BB * NN;

    // ---- output layout: [idx 64][routing 256][attended 4*2048*512] ----
    float* out      = (float*)d_out;
    float* out_idx  = out;
    float* out_rw   = out + BB * 16;
    float* attended = out + BB * 16 + BB * NN;

    // K0: precision conversion
    {
        const int T = 256;
        k_cvt_f32_f16<<<(int)((nXH  + T - 1) / T), T, 0, stream>>>(x,          xh,    (int)nXH);
        k_cvt_f32_f16<<<(int)((nWIN + T - 1) / T), T, 0, stream>>>(in_proj_w,  winh,  (int)nWIN);
        k_cvt_f32_f16<<<(int)((nWOUT+ T - 1) / T), T, 0, stream>>>(out_proj_w, wouth, (int)nWOUT);
        k_cvt_f32_f16<<<(int)((nWC  + T - 1) / T), T, 0, stream>>>(content_w,  wch,   (int)nWC);
    }
    // K1: QKV projection — (8192/16)*(1536/64) = 12288 waves, 4 waves/block
    k_gemm_qkv<<<12288 / 4, 128, 0, stream>>>(xh, winh, in_proj_b, Qh, Kh, Vt);
    // K2: flash attention — 32 (b,h) * 128 q-tiles = 4096 waves
    k_attn<<<4096 / 4, 128, 0, stream>>>(Qh, Kh, Vt, ctxh);
    // K3: out projection — (8192/16)*(512/64) = 4096 waves
    k_gemm_out<<<4096 / 4, 128, 0, stream>>>(ctxh, wouth, out_proj_b, attended);
    // K4: content affinity — 512 waves
    k_gemm_aff<<<512 / 4, 128, 0, stream>>>(attended, wch, content_b, aff);
    // K5: per-(b,n) reductions
    k_reduce<<<BB * NN, 256, 0, stream>>>(aff, sums, maxs);
    // K6: final score + top-16 + one-hot routing
    k_final<<<BB, NN, 0, stream>>>(sums, maxs, out_idx, out_rw);
}
